// BPRMF_4415226380935
// MI455X (gfx1250) — compile-verified, matrix operand order checked by reading, not run on hardware
//
#include <hip/hip_runtime.h>

typedef __attribute__((ext_vector_type(16))) _Float16 v16h;
typedef __attribute__((ext_vector_type(8)))  _Float16 v8h;
typedef __attribute__((ext_vector_type(8)))  float    v8f;
typedef __attribute__((ext_vector_type(4)))  float    v4f;

#define B_      512
#define T_      50
#define D_      128
#define NITEMS  100001
#define VTH     1.0f
#define BN_EPS  1e-5f

// ---------------------------------------------------------------------------
// K1: embedding gather + running-mean prefix.  thread = (b, d)
//   uF[b,t,d] = (sum_{s<=t} emb[seq[b,s], d]) / min(t+1, len[b])
// ---------------------------------------------------------------------------
__global__ void k_uf(const int* __restrict__ seq, const int* __restrict__ len,
                     const float* __restrict__ emb, float* __restrict__ uF) {
  const int tid = blockIdx.x * blockDim.x + threadIdx.x;   // b*128 + d
  const int b = tid >> 7, d = tid & 127;
  const int* srow = seq + b * T_;
  const int L = len[b];
  float prefix = 0.0f;
  float* urow = uF + (size_t)b * (T_ * D_) + d;
  for (int t = 0; t < T_; ++t) {
    const int idx = srow[t];                       // row 0 is the zero pad row
    prefix += emb[(size_t)idx * D_ + d];
    const int dn = (t + 1 < L) ? (t + 1) : L;
    urow[t * D_] = prefix / (float)dn;
  }
}

// ---------------------------------------------------------------------------
// K2: BatchNorm training-mode stats over batch axis.  thread = (t, d)
//   fold into y = uF*scaleA + biasA
// ---------------------------------------------------------------------------
__global__ void k_bn(const float* __restrict__ uF, const float* __restrict__ gamma,
                     const float* __restrict__ beta, float* __restrict__ scaleA,
                     float* __restrict__ biasA) {
  const int tid = blockIdx.x * blockDim.x + threadIdx.x;   // t*128 + d (6400 total)
  const int d = tid & 127;
  float s = 0.0f, ss = 0.0f;
  for (int b = 0; b < B_; ++b) {
    const float v = uF[(size_t)b * (T_ * D_) + tid];
    s += v; ss += v * v;
  }
  const float mean = s * (1.0f / B_);
  const float var  = ss * (1.0f / B_) - mean * mean;       // biased var
  const float sc   = rsqrtf(var + BN_EPS) * gamma[d];
  scaleA[tid] = sc;
  biasA[tid]  = beta[d] - mean * sc;
}

// ---------------------------------------------------------------------------
// K3: LIF recurrence (tau=2, soft reset) + time-mean -> f16 A matrix [512,128]
// ---------------------------------------------------------------------------
__global__ void k_lif(const float* __restrict__ uF, const float* __restrict__ scaleA,
                      const float* __restrict__ biasA, _Float16* __restrict__ A) {
  const int tid = blockIdx.x * blockDim.x + threadIdx.x;   // b*128 + d
  const int b = tid >> 7, d = tid & 127;
  const float* urow = uF + (size_t)b * (T_ * D_) + d;
  float v = 0.0f, acc = 0.0f;
  for (int t = 0; t < T_; ++t) {
    const float x = fmaf(urow[t * D_], scaleA[t * D_ + d], biasA[t * D_ + d]);
    v = 0.5f * (v + x);                         // v += (x - v)/tau, tau = 2
    const float sp = (v >= VTH) ? 1.0f : 0.0f;  // Heaviside(v - v_th)
    v -= sp * VTH;                              // soft reset
    acc += sp;
  }
  A[tid] = (_Float16)(acc * (1.0f / T_));
}

// ---------------------------------------------------------------------------
// K4: one-shot f32 -> f16 conversion of the emb table (L2-resident, read once)
// thread handles 8 consecutive elements.
// ---------------------------------------------------------------------------
#define NE8 ((NITEMS * D_) / 8)    // 12,800,128 / 8 = 1,600,016
__global__ void k_cvt(const float* __restrict__ src, _Float16* __restrict__ dst) {
  const size_t tid = (size_t)blockIdx.x * blockDim.x + threadIdx.x;
  if (tid >= NE8) return;
  const float* p = src + tid * 8;
  const v4f f0 = *(const v4f*)(p);
  const v4f f1 = *(const v4f*)(p + 4);
  const v8f f  = __builtin_shufflevector(f0, f1, 0, 1, 2, 3, 4, 5, 6, 7);
  *(v8h*)(dst + tid * 8) = __builtin_convertvector(f, v8h);
}

// ---------------------------------------------------------------------------
// K5: scores[512, 100001] = A(f16) @ emb16^T(f16), f32 accumulate (WMMA).
// One wave owns a 128(M) x 16(N) strip: 8 accumulators share each B fragment,
// so the 25.6 MB f16 emb table is streamed from L2 only 4x total.
// Fragment layout per CDNA5 ISA (16-bit A/B 16x32):
//   lane l: row = l&15, khalf = l>>4; element j -> K = j + 8*(j>=8) + 8*khalf
// i.e. two contiguous 8-K runs (16 B each) at kb+8*khalf and kb+16+8*khalf.
// Output stores use non-temporal hint: 205 MB write-once stream must not
// evict the L2-resident emb table.
// ---------------------------------------------------------------------------
__global__ void __launch_bounds__(256)
k_gemm(const _Float16* __restrict__ A, const _Float16* __restrict__ emb16,
       float* __restrict__ out) {
  const int lane   = threadIdx.x & 31;
  const int wave   = threadIdx.x >> 5;
  const int n_tile = blockIdx.x * 8 + wave;
  const int m_base = blockIdx.y * 128;

  const int ncol = n_tile * 16 + (lane & 15);
  const int nn   = (ncol < NITEMS) ? ncol : (NITEMS - 1);  // clamp: EXEC stays all-1
  const int kh8  = (lane >> 4) * 8;

  v8f c[8];
#pragma unroll
  for (int mt = 0; mt < 8; ++mt) c[mt] = (v8f){};

#pragma unroll
  for (int kb = 0; kb < D_; kb += 32) {
    // ---- B fragment: emb16 row `nn`, K runs [kb+kh8, +8) and [kb+16+kh8, +8)
    const _Float16* bp = emb16 + (size_t)nn * D_ + kb + kh8;
    const v8h b0 = *(const v8h*)(bp);
    const v8h b1 = *(const v8h*)(bp + 16);
    const v16h bf = __builtin_shufflevector(b0, b1,
        0, 1, 2, 3, 4, 5, 6, 7, 8, 9, 10, 11, 12, 13, 14, 15);

    // ---- A fragments: rows m_base + mt*16 + (lane&15), same K-run pattern
    const _Float16* ap = A + (size_t)(m_base + (lane & 15)) * D_ + kb + kh8;
#pragma unroll
    for (int mt = 0; mt < 8; ++mt) {
      const v8h al = *(const v8h*)(ap + mt * 16 * D_);
      const v8h ah = *(const v8h*)(ap + mt * 16 * D_ + 16);
      const v16h a = __builtin_shufflevector(al, ah,
          0, 1, 2, 3, 4, 5, 6, 7, 8, 9, 10, 11, 12, 13, 14, 15);
      c[mt] = __builtin_amdgcn_wmma_f32_16x16x32_f16(
          false, a, false, bf, (short)0, c[mt], false, false);
    }
  }

  // ---- store: C/D layout M = r + 8*(lane>=16), N = lane&15  (non-temporal)
  if (ncol < NITEMS) {
#pragma unroll
    for (int mt = 0; mt < 8; ++mt) {
#pragma unroll
      for (int r = 0; r < 8; ++r) {
        const int row = m_base + mt * 16 + r + kh8;
        __builtin_nontemporal_store(c[mt][r], out + (size_t)row * NITEMS + ncol);
      }
    }
  }
}

// ---------------------------------------------------------------------------
extern "C" void kernel_launch(void* const* d_in, const int* in_sizes, int n_in,
                              void* d_out, int out_size, void* d_ws, size_t ws_size,
                              hipStream_t stream) {
  const int*   seq   = (const int*)d_in[0];     // [512, 50]
  const int*   len   = (const int*)d_in[1];     // [512]
  const float* emb   = (const float*)d_in[2];   // [100001, 128]
  const float* gamma = (const float*)d_in[3];   // [128]
  const float* beta  = (const float*)d_in[4];   // [128]
  float* out = (float*)d_out;                   // [512, 100001]

  char* ws = (char*)d_ws;
  float*    uF     = (float*)ws;                          // 13,107,200 B
  float*    scaleA = (float*)(ws + 13107200);             //     25,600 B
  float*    biasA  = (float*)(ws + 13132800);             //     25,600 B
  _Float16* Amat   = (_Float16*)(ws + 13158400);          //    131,072 B
  _Float16* emb16  = (_Float16*)(ws + 13289472);          // 25,600,256 B  (tot ~38.9 MB)

  k_uf <<<dim3(256),      dim3(256), 0, stream>>>(seq, len, emb, uF);
  k_bn <<<dim3(25),       dim3(256), 0, stream>>>(uF, gamma, beta, scaleA, biasA);
  k_lif<<<dim3(256),      dim3(256), 0, stream>>>(uF, scaleA, biasA, Amat);
  k_cvt<<<dim3(6251),     dim3(256), 0, stream>>>(emb, emb16);
  // N tiles: ceil(100001/16)=6251 -> 782 blocks of 8 waves; M: 512/128 = 4
  k_gemm<<<dim3(782, 4),  dim3(256), 0, stream>>>(Amat, emb16, out);
}
